// GIN_43550968381727
// MI455X (gfx1250) — matrix operand order repaired and checked by reading, not compiled, not run on hardware
//
#include <hip/hip_runtime.h>
#include <hip/hip_bf16.h>

// ---------------------------------------------------------------------------
// GIN forward on gfx1250: scatter-add aggregation (global_atomic_add_f32) +
// bf16 WMMA GEMMs (v_wmma_f32_16x16x32_bf16), 16x64 register-blocked per wave,
// fused bias+ReLU epilogue with direct bf16 emission for GEMM->GEMM chains.
// ---------------------------------------------------------------------------

#define N_NODES 50000
#define N_EDGES 800000
#define IN_C    128
#define HID_C   256
#define OUT_C   64

typedef __attribute__((ext_vector_type(16))) __bf16 v16bf;
typedef __attribute__((ext_vector_type(8)))  __bf16 v8bf;
typedef __attribute__((ext_vector_type(8)))  float  v8f;

static __device__ __forceinline__ __bf16 f2bf(float f) {
    unsigned u = __builtin_bit_cast(unsigned, f);
    unsigned r = (u + 0x7FFFu + ((u >> 16) & 1u)) >> 16;   // round-to-nearest-even
    unsigned short h = (unsigned short)r;
    return __builtin_bit_cast(__bf16, h);
}

// ---- scatter-add: agg[dst] += feat[src], 4 channels per thread -------------
__global__ void gin_scatter_add(const int* __restrict__ ei,   // [2, N_EDGES]
                                const float* __restrict__ feat,
                                float* __restrict__ agg,
                                int d, int shift, int mask, long long total) {
    long long t = (long long)blockIdx.x * blockDim.x + threadIdx.x;
    if (t >= total) return;
    int e = (int)(t >> shift);
    int c = (int)(t & mask) << 2;                 // channel * 4
    int src = ei[e];
    int dst = ei[N_EDGES + e];
    const float4 v = *reinterpret_cast<const float4*>(feat + (size_t)src * d + c);
    float* a = agg + (size_t)dst * d + c;
    atomicAdd(a + 0, v.x);
    atomicAdd(a + 1, v.y);
    atomicAdd(a + 2, v.z);
    atomicAdd(a + 3, v.w);
}

// ---- f32 -> bf16 elementwise ----------------------------------------------
__global__ void gin_cvt_bf16(__bf16* __restrict__ dst, const float* __restrict__ src,
                             long long n) {
    long long i = (long long)blockIdx.x * blockDim.x + threadIdx.x;
    if (i < n) dst[i] = f2bf(src[i]);
}

// ---- weight transpose + convert: w[K,N] f32 -> wT[N,K] bf16 ---------------
__global__ void gin_wt_prep(__bf16* __restrict__ wT, const float* __restrict__ w,
                            int K, int N) {
    int idx = blockIdx.x * blockDim.x + threadIdx.x;
    if (idx >= K * N) return;
    int k = idx / N;
    int n = idx - k * N;
    wT[(size_t)n * K + k] = f2bf(w[idx]);
}

// ---- WMMA GEMM: out = relu?(A[M,K] @ B + bias), B given as BT[N,K] --------
// blockDim = (32, 4): each 32-lane wave computes a 16x64 output strip
// (4 accumulators; one A fragment reused across 4 WMMAs per K-step).
// Optional f32 and/or bf16 outputs (fused downcast for GEMM->GEMM chains).
__global__ void gin_gemm_bf16(const __bf16* __restrict__ A,
                              const __bf16* __restrict__ BT,
                              const float* __restrict__ bias,
                              float* __restrict__ outF,      // may be null
                              __bf16* __restrict__ outB,     // may be null
                              int M, int K, int N, int relu) {
    int m_tile = blockIdx.y * blockDim.y + threadIdx.y;
    if (m_tile >= (M >> 4)) return;               // wave-uniform guard
    int lane = threadIdx.x;                       // 0..31 (wave32)
    int half = lane >> 4;                         // 0: lanes 0-15, 1: lanes 16-31
    int l15  = lane & 15;

    int m_base = m_tile << 4;
    int n_base = blockIdx.x << 6;                 // 64 columns per wave

    const __bf16* Arow = A + (size_t)(m_base + l15) * K;
    const __bf16* Bcol[4];
    #pragma unroll
    for (int j = 0; j < 4; ++j)
        Bcol[j] = BT + (size_t)(n_base + j * 16 + l15) * K + half * 16;

    v8f acc[4] = {};
    for (int k0 = 0; k0 < K; k0 += 32) {
        // A fragment (16x32 bf16): lane<16 -> K 0..15; lane>=16 -> 16..31
        v8bf lo = *reinterpret_cast<const v8bf*>(Arow + k0 + half * 16);
        v8bf hi = *reinterpret_cast<const v8bf*>(Arow + k0 + half * 16 + 8);
        v16bf a = __builtin_shufflevector(lo, hi, 0,1,2,3,4,5,6,7,8,9,10,11,12,13,14,15);
        #pragma unroll
        for (int j = 0; j < 4; ++j) {
            // B fragment (32x16 bf16): lane = column N, K 0..15 / 16..31 per half-wave
            v16bf b = *reinterpret_cast<const v16bf*>(Bcol[j] + k0);
            acc[j] = __builtin_amdgcn_wmma_f32_16x16x32_bf16(
                /*neg_a=*/false, a, /*neg_b=*/false, b,
                /*c_mod=*/(short)0, acc[j], /*reuse_a=*/false, /*reuse_b=*/false);
        }
    }

    // Epilogue: D vgpr i -> row m_base + i + 8*half, col = n_base + j*16 + l15
    size_t rbase = (size_t)(m_base + half * 8) * N;
    #pragma unroll
    for (int j = 0; j < 4; ++j) {
        int col = n_base + j * 16 + l15;
        float bv = bias[col];
        #pragma unroll
        for (int i = 0; i < 8; ++i) {
            float v = acc[j][i] + bv;
            if (relu) v = fmaxf(v, 0.0f);
            size_t off = rbase + (size_t)i * N + col;
            if (outF) outF[off] = v;
            if (outB) outB[off] = f2bf(v);
        }
    }
}

// ---------------------------------------------------------------------------

static inline void launch_gemm(const __bf16* A, const __bf16* BT, const float* bias,
                               float* outF, __bf16* outB,
                               int M, int K, int N, int relu, hipStream_t stream) {
    dim3 block(32, 4);
    dim3 grid(N / 64, (M / 16 + 3) / 4);
    gin_gemm_bf16<<<grid, block, 0, stream>>>(A, BT, bias, outF, outB, M, K, N, relu);
}

static inline void launch_scatter(const int* ei, const float* feat, float* agg,
                                  int d, hipStream_t stream) {
    int shift = (d == 128) ? 5 : 6;               // log2(d/4)
    int mask  = (d >> 2) - 1;
    long long total = (long long)N_EDGES * (d >> 2);
    long long blocks = (total + 255) / 256;
    gin_scatter_add<<<(unsigned)blocks, 256, 0, stream>>>(ei, feat, agg, d, shift, mask, total);
}

static inline void launch_cvt(__bf16* dst, const float* src, long long n, hipStream_t stream) {
    long long blocks = (n + 255) / 256;
    gin_cvt_bf16<<<(unsigned)blocks, 256, 0, stream>>>(dst, src, n);
}

extern "C" void kernel_launch(void* const* d_in, const int* in_sizes, int n_in,
                              void* d_out, int out_size, void* d_ws, size_t ws_size,
                              hipStream_t stream) {
    const float* x   = (const float*)d_in[0];
    const int*   ei  = (const int*)d_in[1];
    const float* w1a = (const float*)d_in[2];  const float* b1a = (const float*)d_in[3];
    const float* w1b = (const float*)d_in[4];  const float* b1b = (const float*)d_in[5];
    const float* w2a = (const float*)d_in[6];  const float* b2a = (const float*)d_in[7];
    const float* w2b = (const float*)d_in[8];  const float* b2b = (const float*)d_in[9];
    const float* w3a = (const float*)d_in[10]; const float* b3a = (const float*)d_in[11];
    const float* w3b = (const float*)d_in[12]; const float* b3b = (const float*)d_in[13];
    float* out = (float*)d_out;

    // ---- workspace layout ----
    char* ws = (char*)d_ws;
    float*  hA   = (float*)(ws + 0);                          // 50000*256 f32
    float*  hB   = (float*)(ws + 51200000);                   // 50000*256 f32
    __bf16* actA = (__bf16*)(ws + 102400000);                 // 50000*256 bf16
    __bf16* actB = (__bf16*)(ws + 128000000);                 // 50000*256 bf16
    __bf16* wT   = (__bf16*)(ws + 153600000);                 // 311296 bf16
    __bf16* w1aT = wT;            // [256,128]
    __bf16* w1bT = wT + 32768;    // [256,256]
    __bf16* w2aT = wT + 98304;    // [256,256]
    __bf16* w2bT = wT + 163840;   // [256,256]
    __bf16* w3aT = wT + 229376;   // [256,256]
    __bf16* w3bT = wT + 294912;   // [64,256]

    // ---- weight prep (transpose + bf16) ----
    gin_wt_prep<<<(IN_C*HID_C + 255)/256, 256, 0, stream>>>(w1aT, w1a, IN_C,  HID_C);
    gin_wt_prep<<<(HID_C*HID_C + 255)/256, 256, 0, stream>>>(w1bT, w1b, HID_C, HID_C);
    gin_wt_prep<<<(HID_C*HID_C + 255)/256, 256, 0, stream>>>(w2aT, w2a, HID_C, HID_C);
    gin_wt_prep<<<(HID_C*HID_C + 255)/256, 256, 0, stream>>>(w2bT, w2b, HID_C, HID_C);
    gin_wt_prep<<<(HID_C*HID_C + 255)/256, 256, 0, stream>>>(w3aT, w3a, HID_C, HID_C);
    gin_wt_prep<<<(HID_C*OUT_C + 255)/256, 256, 0, stream>>>(w3bT, w3b, HID_C, OUT_C);

    // ---- layer 1: h1 = relu( relu((x + agg) @ w1a + b1a) @ w1b + b1b ) ----
    hipMemcpyAsync(hB, x, (size_t)N_NODES * IN_C * sizeof(float),
                   hipMemcpyDeviceToDevice, stream);          // self term
    launch_scatter(ei, x, hB, IN_C, stream);                  // hB = x + segment_sum
    launch_cvt(actA, hB, (long long)N_NODES * IN_C, stream);
    launch_gemm(actA, w1aT, b1a, nullptr, actB, N_NODES, IN_C,  HID_C, 1, stream);
    launch_gemm(actB, w1bT, b1b, hA,      nullptr, N_NODES, HID_C, HID_C, 1, stream);

    // ---- layer 2 ----
    hipMemcpyAsync(hB, hA, (size_t)N_NODES * HID_C * sizeof(float),
                   hipMemcpyDeviceToDevice, stream);
    launch_scatter(ei, hA, hB, HID_C, stream);
    launch_cvt(actA, hB, (long long)N_NODES * HID_C, stream);
    launch_gemm(actA, w2aT, b2a, nullptr, actB, N_NODES, HID_C, HID_C, 1, stream);
    launch_gemm(actB, w2bT, b2b, nullptr, actA, N_NODES, HID_C, HID_C, 1, stream);

    // ---- head: relu(h2 @ w3a + b3a) @ w3b + b3b ----
    launch_gemm(actA, w3aT, b3a, nullptr, actB, N_NODES, HID_C, HID_C, 1, stream);
    launch_gemm(actB, w3bT, b3b, out,     nullptr, N_NODES, HID_C, OUT_C, 0, stream);
}
